// GNNLayer_43456479101183
// MI455X (gfx1250) — matrix operand order repaired and checked by reading, not compiled, not run on hardware
//
#include <hip/hip_runtime.h>
#include <math.h>

// Problem constants (match reference)
#define NR 20000      // rows
#define NT 100000     // tokens
#define NE 300000     // edges per direction
#define DD 256        // hidden dim
#define DE 128        // edge dim
#define HH 4          // heads
#define SCALE_F (0.125f)   // 1/sqrt(64)
#define EPS_F (1e-5f)

typedef __attribute__((ext_vector_type(16))) __bf16 v16bf;
typedef __attribute__((ext_vector_type(8)))  float  v8f;
typedef __attribute__((ext_vector_type(8)))  unsigned int v8u;

__device__ __forceinline__ unsigned short f32_to_bf16_bits(float f) {
    unsigned u = __float_as_uint(f);
    u += 0x7fffu + ((u >> 16) & 1u);   // round-to-nearest-even
    return (unsigned short)(u >> 16);
}

// ---------------------------------------------------------------------------
// f32 -> bf16 conversion (grid-stride), row-major preserved (activations)
// ---------------------------------------------------------------------------
__global__ void f32_to_bf16_kernel(const float* __restrict__ in,
                                   unsigned short* __restrict__ out, size_t n) {
    size_t i = (size_t)blockIdx.x * blockDim.x + threadIdx.x;
    size_t stride = (size_t)gridDim.x * blockDim.x;
    for (; i < n; i += stride) out[i] = f32_to_bf16_bits(in[i]);
}

// f32 [K,N] -> bf16 [N,K] (transposed weights; coalesced read)
__global__ void f32_to_bf16_T_kernel(const float* __restrict__ in,
                                     unsigned short* __restrict__ out,
                                     int K, int N) {
    int i = blockIdx.x * blockDim.x + threadIdx.x;
    int total = K * N;
    if (i >= total) return;
    int k = i / N, n = i - k * N;
    out[(size_t)n * K + k] = f32_to_bf16_bits(in[i]);
}

__global__ void fill_f32_kernel(float* __restrict__ p, float v, size_t n) {
    size_t i = (size_t)blockIdx.x * blockDim.x + threadIdx.x;
    size_t stride = (size_t)gridDim.x * blockDim.x;
    for (; i < n; i += stride) p[i] = v;
}

// ---------------------------------------------------------------------------
// WMMA GEMM:  C[M,N] = bf16(A[M,K]) @ B + bias[N], B given transposed bf16 [N,K].
// block = 128 threads (4 waves); each wave -> 16x32 strip (two 16x16 tiles,
// A reused). block -> 16x128. grid = (ceil(M/16), N/128). K templated so the
// K-loop fully unrolls (pure global_load_b128 + v_wmma stream).
// ---------------------------------------------------------------------------
template<int K>
__global__ __launch_bounds__(128)
void wmma_gemm_bias_kernel(const unsigned short* __restrict__ A,
                           const unsigned short* __restrict__ Bt,
                           const float* __restrict__ bias,
                           float* __restrict__ Cmat,
                           int M, int N) {
    const int lane = threadIdx.x & 31;
    const int wave = threadIdx.x >> 5;
    const int lo = lane & 15;
    const int hi = lane >> 4;            // 0 or 1

    const int m0 = blockIdx.x * 16;
    const int n0 = blockIdx.y * 128 + wave * 32;

    // Clamp out-of-range A rows instead of zeroing: their accumulators are
    // garbage but never stored, so no per-element masking is needed.
    const int rowA = m0 + lo;
    const int rowClp = rowA < M ? rowA : (M - 1);

    const uint4* Ap  = (const uint4*)(A  + (size_t)rowClp * K + hi * 16);
    const uint4* Bp0 = (const uint4*)(Bt + (size_t)(n0 + lo) * K + hi * 16);
    const uint4* Bp1 = (const uint4*)(Bt + (size_t)(n0 + 16 + lo) * K + hi * 16);

    v8f acc0 = {}, acc1 = {};
#pragma unroll
    for (int ks = 0; ks < K / 32; ++ks) {
        uint4 aL = Ap[ks * 4];        // a[0..7]  : K = ks*32 + hi*16 + 0..7
        uint4 aH = Ap[ks * 4 + 1];    // a[8..15] : K = ks*32 + hi*16 + 8..15
        uint4 b0L = Bp0[ks * 4];      // b[0..7]  : K = ks*32 + hi*16 + 0..7
        uint4 b0H = Bp0[ks * 4 + 1];  // b[8..15] : K = ks*32 + hi*16 + 8..15
        uint4 b1L = Bp1[ks * 4];
        uint4 b1H = Bp1[ks * 4 + 1];

        v8u au = {aL.x, aL.y, aL.z, aL.w, aH.x, aH.y, aH.z, aH.w};
        v8u b0u = {b0L.x, b0L.y, b0L.z, b0L.w, b0H.x, b0H.y, b0H.z, b0H.w};
        v8u b1u = {b1L.x, b1L.y, b1L.z, b1L.w, b1H.x, b1H.y, b1H.z, b1H.w};
        v16bf a  = __builtin_bit_cast(v16bf, au);
        v16bf b0 = __builtin_bit_cast(v16bf, b0u);
        v16bf b1 = __builtin_bit_cast(v16bf, b1u);

        acc0 = __builtin_amdgcn_wmma_f32_16x16x32_bf16(
            false, a, false, b0, (short)0, acc0, false, false);
        acc1 = __builtin_amdgcn_wmma_f32_16x16x32_bf16(
            false, a, false, b1, (short)0, acc1, false, false);
    }

    const float bv0 = bias[n0 + lo];
    const float bv1 = bias[n0 + 16 + lo];
    float* cp = Cmat + (size_t)(m0 + hi * 8) * N + n0 + lo;
    if (m0 + 16 <= M) {               // uniform fast path (all full tiles)
#pragma unroll
        for (int r = 0; r < 8; ++r) {
            cp[(size_t)r * N]      = acc0[r] + bv0;
            cp[(size_t)r * N + 16] = acc1[r] + bv1;
        }
    } else {
#pragma unroll
        for (int r = 0; r < 8; ++r) {
            if (m0 + r + hi * 8 < M) {
                cp[(size_t)r * N]      = acc0[r] + bv0;
                cp[(size_t)r * N + 16] = acc1[r] + bv1;
            }
        }
    }
}

// ---------------------------------------------------------------------------
// Edge pass 1: logits[e,h] = SCALE * dot(q[dst,h,:], k[src,h,:] + eproj[e,h,:])
// plus segment max into amax[dst,h].   One wave per edge, lane handles 8 ch.
// ---------------------------------------------------------------------------
__device__ __forceinline__ void atomicMaxF(float* addr, float val) {
    if (val >= 0.f) atomicMax((int*)addr, __float_as_int(val));
    else            atomicMin((unsigned int*)addr, __float_as_uint(val));
}

__global__ __launch_bounds__(256)
void edge_logits_kernel(const int* __restrict__ ei,
                        const float* __restrict__ q,
                        const float* __restrict__ kmat,
                        const float* __restrict__ emat,
                        float* __restrict__ attn,
                        float* __restrict__ amax, int E) {
    int e = blockIdx.x * 8 + (threadIdx.x >> 5);
    if (e >= E) return;
    int lane = threadIdx.x & 31;
    int s = ei[e];
    int d = ei[E + e];
    const float4* qp = (const float4*)(q    + (size_t)d * DD);
    const float4* kp = (const float4*)(kmat + (size_t)s * DD);
    const float4* ep = (const float4*)(emat + (size_t)e * DD);
    float4 q0 = qp[lane * 2], q1 = qp[lane * 2 + 1];
    float4 k0 = kp[lane * 2], k1 = kp[lane * 2 + 1];
    float4 e0 = ep[lane * 2], e1 = ep[lane * 2 + 1];
    float p = q0.x * (k0.x + e0.x) + q0.y * (k0.y + e0.y)
            + q0.z * (k0.z + e0.z) + q0.w * (k0.w + e0.w)
            + q1.x * (k1.x + e1.x) + q1.y * (k1.y + e1.y)
            + q1.z * (k1.z + e1.z) + q1.w * (k1.w + e1.w);
    p += __shfl_xor(p, 1, 32);
    p += __shfl_xor(p, 2, 32);
    p += __shfl_xor(p, 4, 32);
    if ((lane & 7) == 0) {
        int h = lane >> 3;
        float lg = p * SCALE_F;
        attn[(size_t)e * HH + h] = lg;
        atomicMaxF(&amax[(size_t)d * HH + h], lg);
    }
}

// ---------------------------------------------------------------------------
// Edge pass 2: ex = exp(logit - amax[dst]); attn <- ex; denom[dst,h] += ex
// ---------------------------------------------------------------------------
__global__ __launch_bounds__(256)
void edge_softmax_kernel(const int* __restrict__ ei,
                         float* __restrict__ attn,
                         const float* __restrict__ amax,
                         float* __restrict__ denom, int E) {
    int i = blockIdx.x * blockDim.x + threadIdx.x;
    if (i >= E * HH) return;
    int e = i >> 2, h = i & 3;
    int d = ei[E + e];
    float ex = __expf(attn[i] - amax[(size_t)d * HH + h]);
    attn[i] = ex;
    atomicAdd(&denom[(size_t)d * HH + h], ex);
}

// ---------------------------------------------------------------------------
// Edge pass 3: out[dst,:] += alpha[e,h] * (v[src,:] + eproj[e,:])
// One wave per edge, 8 channels/lane, f32 global atomics.
// ---------------------------------------------------------------------------
__global__ __launch_bounds__(256)
void edge_scatter_kernel(const int* __restrict__ ei,
                         const float* __restrict__ attn,
                         const float* __restrict__ denom,
                         const float* __restrict__ vmat,
                         const float* __restrict__ emat,
                         float* __restrict__ out, int E) {
    int e = blockIdx.x * 8 + (threadIdx.x >> 5);
    if (e >= E) return;
    int lane = threadIdx.x & 31;
    int s = ei[e];
    int d = ei[E + e];
    int h = lane >> 3;
    float alpha = attn[(size_t)e * HH + h] / denom[(size_t)d * HH + h];
    const float4* vp = (const float4*)(vmat + (size_t)s * DD);
    const float4* ep = (const float4*)(emat + (size_t)e * DD);
    float4 v0 = vp[lane * 2], v1 = vp[lane * 2 + 1];
    float4 e0 = ep[lane * 2], e1 = ep[lane * 2 + 1];
    float* op = out + (size_t)d * DD + lane * 8;
    atomicAdd(op + 0, (v0.x + e0.x) * alpha);
    atomicAdd(op + 1, (v0.y + e0.y) * alpha);
    atomicAdd(op + 2, (v0.z + e0.z) * alpha);
    atomicAdd(op + 3, (v0.w + e0.w) * alpha);
    atomicAdd(op + 4, (v1.x + e1.x) * alpha);
    atomicAdd(op + 5, (v1.y + e1.y) * alpha);
    atomicAdd(op + 6, (v1.z + e1.z) * alpha);
    atomicAdd(op + 7, (v1.w + e1.w) * alpha);
}

// ---------------------------------------------------------------------------
// Fused residual + LayerNorm. One wave per 256-wide row. Optionally emits a
// bf16 copy (needed as the next conv's source activations).
// ---------------------------------------------------------------------------
__global__ __launch_bounds__(256)
void add_ln_kernel(const float* __restrict__ x, const float* __restrict__ msg,
                   const float* __restrict__ gamma, const float* __restrict__ beta,
                   float* __restrict__ out, unsigned short* __restrict__ out_bf,
                   int Nrows) {
    int row = blockIdx.x * 8 + (threadIdx.x >> 5);
    if (row >= Nrows) return;
    int lane = threadIdx.x & 31;
    const float* xr = x   + (size_t)row * DD;
    const float* mr = msg + (size_t)row * DD;
    float t[8];
    float sum = 0.f;
#pragma unroll
    for (int i = 0; i < 8; ++i) {
        int c = lane + 32 * i;
        t[i] = xr[c] + mr[c];
        sum += t[i];
    }
#pragma unroll
    for (int off = 1; off < 32; off <<= 1) sum += __shfl_xor(sum, off, 32);
    float mean = sum * (1.f / 256.f);
    float vs = 0.f;
#pragma unroll
    for (int i = 0; i < 8; ++i) { float dlt = t[i] - mean; vs += dlt * dlt; }
#pragma unroll
    for (int off = 1; off < 32; off <<= 1) vs += __shfl_xor(vs, off, 32);
    float rstd = rsqrtf(vs * (1.f / 256.f) + EPS_F);
#pragma unroll
    for (int i = 0; i < 8; ++i) {
        int c = lane + 32 * i;
        float o = (t[i] - mean) * rstd * gamma[c] + beta[c];
        out[(size_t)row * DD + c] = o;
        if (out_bf) out_bf[(size_t)row * DD + c] = f32_to_bf16_bits(o);
    }
}

// ---------------------------------------------------------------------------
// Host orchestration
// ---------------------------------------------------------------------------
static char* carve(char*& p, size_t bytes) {
    char* r = p;
    p += (bytes + 255) & ~(size_t)255;
    return r;
}

extern "C" void kernel_launch(void* const* d_in, const int* in_sizes, int n_in,
                              void* d_out, int out_size, void* d_ws, size_t ws_size,
                              hipStream_t stream) {
    (void)in_sizes; (void)n_in; (void)out_size; (void)ws_size;

    const float* row_x   = (const float*)d_in[0];
    const float* token_x = (const float*)d_in[1];
    const int*   t2r_ei  = (const int*)d_in[2];
    const float* ea_t2r  = (const float*)d_in[3];
    const int*   r2t_ei  = (const int*)d_in[4];
    const float* ea_r2t  = (const float*)d_in[5];
    // t2r params: Wq,bq,Wk,bk,Wv,bv,We,be,Ws,bs  -> indices 6..15
    // r2t params: same                            -> indices 16..25
    const float* row_gamma   = (const float*)d_in[26];
    const float* row_beta    = (const float*)d_in[27];
    const float* token_gamma = (const float*)d_in[28];
    const float* token_beta  = (const float*)d_in[29];

    float* out_row = (float*)d_out;                       // [NR, 256]
    float* out_tok = (float*)d_out + (size_t)NR * DD;     // [NT, 256]

    // ---- workspace layout -------------------------------------------------
    char* p = (char*)d_ws;
    float* qbuf   = (float*)carve(p, (size_t)NT * DD * 4);
    float* kbuf   = (float*)carve(p, (size_t)NT * DD * 4);
    float* vbuf   = (float*)carve(p, (size_t)NT * DD * 4);
    float* ebuf   = (float*)carve(p, (size_t)NE * DD * 4);
    float* msgbuf = (float*)carve(p, (size_t)NT * DD * 4);
    float* attn   = (float*)carve(p, (size_t)NE * HH * 4);
    float* amax   = (float*)carve(p, (size_t)NT * HH * 4);
    float* denom  = (float*)carve(p, (size_t)NT * HH * 4);
    unsigned short* rowx_bf   = (unsigned short*)carve(p, (size_t)NR * DD * 2);
    unsigned short* tokx_bf   = (unsigned short*)carve(p, (size_t)NT * DD * 2);
    unsigned short* rowout_bf = (unsigned short*)carve(p, (size_t)NR * DD * 2);
    unsigned short* ea1_bf    = (unsigned short*)carve(p, (size_t)NE * DE * 2);
    unsigned short* ea2_bf    = (unsigned short*)carve(p, (size_t)NE * DE * 2);
    unsigned short* wbf[10];   // transposed bf16 weights [N=256, K]
    for (int i = 0; i < 10; ++i) wbf[i] = (unsigned short*)carve(p, (size_t)DD * DD * 2);

    auto cvt = [&](const float* src, unsigned short* dst, size_t n) {
        f32_to_bf16_kernel<<<1024, 256, 0, stream>>>(src, dst, n);
    };
    auto cvtT = [&](const float* src, unsigned short* dst, int K) {
        f32_to_bf16_T_kernel<<<(K * DD + 255) / 256, 256, 0, stream>>>(src, dst, K, DD);
    };
    auto fill = [&](float* dst, float v, size_t n) {
        fill_f32_kernel<<<512, 256, 0, stream>>>(dst, v, n);
    };
    auto gemm256 = [&](const unsigned short* A, const unsigned short* Bt,
                       const float* bias, float* C, int M) {
        dim3 grid((M + 15) / 16, DD / 128);
        wmma_gemm_bias_kernel<256><<<grid, 128, 0, stream>>>(A, Bt, bias, C, M, DD);
    };
    auto gemm128 = [&](const unsigned short* A, const unsigned short* Bt,
                       const float* bias, float* C, int M) {
        dim3 grid((M + 15) / 16, DD / 128);
        wmma_gemm_bias_kernel<128><<<grid, 128, 0, stream>>>(A, Bt, bias, C, M, DD);
    };

    // ---- precision conversions --------------------------------------------
    cvt(row_x,   rowx_bf, (size_t)NR * DD);
    cvt(token_x, tokx_bf, (size_t)NT * DD);
    cvt(ea_t2r,  ea1_bf,  (size_t)NE * DE);
    cvt(ea_r2t,  ea2_bf,  (size_t)NE * DE);
    // weights (transposed to [N,K] bf16): t2r Wq,Wk,Wv,We,Ws -> wbf[0..4]; r2t -> wbf[5..9]
    const int widx[5] = {6, 8, 10, 12, 14};
    for (int i = 0; i < 5; ++i) {
        int kdim = (i == 3) ? DE : DD;   // We is [128,256]
        cvtT((const float*)d_in[widx[i]],      wbf[i],     kdim);
        cvtT((const float*)d_in[widx[i] + 10], wbf[i + 5], kdim);
    }

    const int eblocks = (NE + 7) / 8;
    const int sblocks = (NE * HH + 255) / 256;

    // ======================= Phase A: t2r (token -> row) ===================
    gemm256(rowx_bf, wbf[0], (const float*)d_in[7],  qbuf,   NR);  // q (dst=row)
    gemm256(tokx_bf, wbf[1], (const float*)d_in[9],  kbuf,   NT);  // k (src=token)
    gemm256(tokx_bf, wbf[2], (const float*)d_in[11], vbuf,   NT);  // v
    gemm128(ea1_bf,  wbf[3], (const float*)d_in[13], ebuf,   NE);  // edge proj
    gemm256(rowx_bf, wbf[4], (const float*)d_in[15], msgbuf, NR);  // skip lin -> msg init

    fill(amax,  -INFINITY, (size_t)NR * HH);
    fill(denom, 0.f,       (size_t)NR * HH);

    edge_logits_kernel <<<eblocks, 256, 0, stream>>>(t2r_ei, qbuf, kbuf, ebuf, attn, amax, NE);
    edge_softmax_kernel<<<sblocks, 256, 0, stream>>>(t2r_ei, attn, amax, denom, NE);
    edge_scatter_kernel<<<eblocks, 256, 0, stream>>>(t2r_ei, attn, denom, vbuf, ebuf, msgbuf, NE);

    add_ln_kernel<<<(NR + 7) / 8, 256, 0, stream>>>(
        row_x, msgbuf, row_gamma, row_beta, out_row, rowout_bf, NR);

    // ======================= Phase B: r2t (row_out -> token) ===============
    gemm256(tokx_bf,   wbf[5], (const float*)d_in[17], qbuf,   NT); // q (dst=token)
    gemm256(rowout_bf, wbf[6], (const float*)d_in[19], kbuf,   NR); // k (src=row_out)
    gemm256(rowout_bf, wbf[7], (const float*)d_in[21], vbuf,   NR); // v
    gemm128(ea2_bf,    wbf[8], (const float*)d_in[23], ebuf,   NE); // edge proj
    gemm256(tokx_bf,   wbf[9], (const float*)d_in[25], msgbuf, NT); // skip lin

    fill(amax,  -INFINITY, (size_t)NT * HH);
    fill(denom, 0.f,       (size_t)NT * HH);

    edge_logits_kernel <<<eblocks, 256, 0, stream>>>(r2t_ei, qbuf, kbuf, ebuf, attn, amax, NE);
    edge_softmax_kernel<<<sblocks, 256, 0, stream>>>(r2t_ei, attn, amax, denom, NE);
    edge_scatter_kernel<<<eblocks, 256, 0, stream>>>(r2t_ei, attn, denom, vbuf, ebuf, msgbuf, NE);

    add_ln_kernel<<<(NT + 7) / 8, 256, 0, stream>>>(
        token_x, msgbuf, token_gamma, token_beta, out_tok, (unsigned short*)nullptr, NT);
}
